// SparseBlockAttention_25374666785276
// MI455X (gfx1250) — matrix operand order repaired and checked by reading, not compiled, not run on hardware
//
#include <hip/hip_runtime.h>
#include <hip/hip_bf16.h>
#include <math.h>

typedef __bf16 bf16;
typedef __attribute__((ext_vector_type(16))) __bf16 v16bf;
typedef __attribute__((ext_vector_type(4)))  __bf16 v4bf;
typedef __attribute__((ext_vector_type(8)))  float   v8f;
typedef __attribute__((ext_vector_type(4)))  float   f32x4;
typedef __attribute__((ext_vector_type(4)))  int     v4i;

#define D_MODEL  2048
#define N_HEADS  16
#define HEAD_DIM 128
#define BB       2
#define SS       2048

#ifndef __has_builtin
#define __has_builtin(x) 0
#endif

// gfx1250 async memory->LDS path (ASYNCcnt-tracked), with safe fallback.
#if __has_builtin(__builtin_amdgcn_global_load_async_to_lds_b128)
#define HAVE_ASYNC_LDS 1
#else
#define HAVE_ASYNC_LDS 0
#endif

#if HAVE_ASYNC_LDS
// builtin signature: (v4i addrspace(1)* src, v4i addrspace(3)* dst, Ii, Ii)
typedef __attribute__((address_space(1))) v4i as1_v4i;
typedef __attribute__((address_space(3))) v4i as3_v4i;
// one 16-byte lane-wise DMA: LDS[dst] = MEM[src]  (global_load_async_to_lds_b128)
__device__ __forceinline__ void async_cp16(bf16* lds_dst, const bf16* g_src) {
  __builtin_amdgcn_global_load_async_to_lds_b128(
      (as1_v4i*)g_src, (as3_v4i*)lds_dst, /*offset=*/0, /*cpol=*/0);
}
__device__ __forceinline__ void wait_async0() {
#if __has_builtin(__builtin_amdgcn_s_wait_asynccnt)
  __builtin_amdgcn_s_wait_asynccnt(0);
#else
  asm volatile("s_wait_asynccnt 0x0" ::: "memory");
#endif
}
#endif

// ---------------------------------------------------------------------------
// WMMA helper: D = A(16x32 bf16) * B(32x16 bf16) + C(16x16 f32)
// ---------------------------------------------------------------------------
__device__ __forceinline__ v8f wmma_bf16(v16bf a, v16bf b, v8f c) {
  return __builtin_amdgcn_wmma_f32_16x16x32_bf16(
      /*neg_a=*/false, a, /*neg_b=*/false, b,
      /*c_mod=*/(short)0, c, /*reuse_a=*/false, /*reuse_b=*/false);
}

// Build a 16-element bf16 fragment from two 16-byte-aligned 8-element halves.
__device__ __forceinline__ v16bf ld_frag(const bf16* p0, const bf16* p1) {
  union { v16bf v; f32x4 h[2]; } u;
  u.h[0] = *(const f32x4*)p0;
  u.h[1] = *(const f32x4*)p1;
  return u.v;
}

// ---------------------------------------------------------------------------
// fp32 -> bf16 conversion (vectorized x4)
// ---------------------------------------------------------------------------
__global__ void cvt_f32_bf16(const float* __restrict__ src,
                             bf16* __restrict__ dst, int n) {
  int i = (blockIdx.x * blockDim.x + threadIdx.x) * 4;
  if (i + 3 < n) {
    f32x4 v = *(const f32x4*)(src + i);
    v4bf o;
    o.x = (bf16)v.x; o.y = (bf16)v.y; o.z = (bf16)v.z; o.w = (bf16)v.w;
    *(v4bf*)(dst + i) = o;
  }
}

// ---------------------------------------------------------------------------
// In-place RoPE on bf16 activations laid out [B*S, N_HEADS*HEAD_DIM].
// ---------------------------------------------------------------------------
__global__ void rope_kernel(bf16* __restrict__ X, const int* __restrict__ pos,
                            int total) {
  int idx = blockIdx.x * blockDim.x + threadIdx.x;
  if (idx >= total) return;
  int i  = idx & 63;
  int hh = (idx >> 6) & (N_HEADS - 1);
  int t  = idx >> 10;                       // b*S + s
  float inv = __expf(-(float)i * 0.14391156856f);  // 10000^(-i/64)
  float ang = (float)pos[t] * inv;
  float sn, cs;
  __sincosf(ang, &sn, &cs);
  bf16* p = X + (size_t)t * D_MODEL + hh * HEAD_DIM;
  float x1 = (float)p[i];
  float x2 = (float)p[i + 64];
  p[i]      = (bf16)(x1 * cs - x2 * sn);
  p[i + 64] = (bf16)(x2 * cs + x1 * sn);
}

// ---------------------------------------------------------------------------
// bf16 WMMA GEMM: C[M,N] = A[M,K] * W[K,N]
// block = 128 threads (4 waves), 64(M) x 128(N) block tile, K-chunk = 32.
// Each wave owns a 32x64 register tile (2x4 WMMA accumulators): 32 WMMAs
// per block per chunk against 12KB of LDS staging.
// ---------------------------------------------------------------------------
template <bool OUT_F32>
__global__ __launch_bounds__(128) void gemm_bf16k(
    const bf16* __restrict__ A, const bf16* __restrict__ W,
    void* __restrict__ Cout, int M, int N, int K) {
  __shared__ __align__(32) bf16 sA[64 * 32];    // A tile, row-major [m][k]
  __shared__ __align__(32) bf16 sBt[128 * 32];  // W tile transposed [n][k]

  const int tid  = threadIdx.x;
  const int lane = tid & 31;
  const int wid  = tid >> 5;
  const int li   = lane & 15;
  const int hi   = lane >> 4;
  const int m0   = blockIdx.y * 64;
  const int n0   = blockIdx.x * 128;
  const int wm   = (wid >> 1) * 32;
  const int wn   = (wid & 1) * 64;

  // cooperative-load assignments
  const int ar  = tid >> 1;          // A row 0..63
  const int ac  = (tid & 1) * 16;    // A col 0/16
  const int brk = tid >> 2;          // W row (k) 0..31
  const int bcn = (tid & 3) * 32;    // W col 0..96

  v8f acc[2][4] = {};

  for (int kc = 0; kc < K; kc += 32) {
    __syncthreads();
    // stage A 64x32 (contiguous 16B pieces -> async DMA when available)
    {
      const bf16* g = A + (size_t)(m0 + ar) * K + kc + ac;
#if HAVE_ASYNC_LDS
      async_cp16(&sA[ar * 32 + ac], g);
      async_cp16(&sA[ar * 32 + ac + 8], g + 8);
#else
      *(f32x4*)&sA[ar * 32 + ac]     = *(const f32x4*)g;
      *(f32x4*)&sA[ar * 32 + ac + 8] = *(const f32x4*)(g + 8);
#endif
    }
    // stage W 32x128 transposed -> sBt[n][k] (element transpose, via VGPRs)
    {
      const bf16* g = W + (size_t)(kc + brk) * N + n0 + bcn;
      union { f32x4 f[4]; bf16 e[32]; } u;
#pragma unroll
      for (int j = 0; j < 4; ++j) u.f[j] = *(const f32x4*)(g + j * 8);
#pragma unroll
      for (int j = 0; j < 32; ++j) sBt[(bcn + j) * 32 + brk] = u.e[j];
    }
    if (kc + 32 < K) {  // prefetch next chunk (global_prefetch_b8)
      __builtin_prefetch(A + (size_t)(m0 + ar) * K + kc + 32 + ac, 0, 1);
      __builtin_prefetch(W + (size_t)(kc + 32 + brk) * N + n0 + bcn, 0, 1);
    }
#if HAVE_ASYNC_LDS
    wait_async0();
#endif
    __syncthreads();

    v16bf af[2], bfrag[4];
#pragma unroll
    for (int t = 0; t < 2; ++t) {
      int ml = wm + t * 16 + li;   // A row
      af[t] = ld_frag(&sA[ml * 32 + hi * 16], &sA[ml * 32 + hi * 16 + 8]);
    }
#pragma unroll
    for (int t = 0; t < 4; ++t) {
      int nl = wn + t * 16 + li;   // B col (row of sBt)
      bfrag[t] = *(const v16bf*)&sBt[nl * 32 + hi * 16];
    }
#pragma unroll
    for (int tm = 0; tm < 2; ++tm)
#pragma unroll
      for (int tn = 0; tn < 4; ++tn)
        acc[tm][tn] = wmma_bf16(af[tm], bfrag[tn], acc[tm][tn]);
  }

  // epilogue: C layout -> lane holds col = li, rows v + 8*hi
#pragma unroll
  for (int tm = 0; tm < 2; ++tm)
#pragma unroll
    for (int tn = 0; tn < 4; ++tn)
#pragma unroll
      for (int v = 0; v < 8; ++v) {
        int m = m0 + wm + tm * 16 + v + 8 * hi;
        int n = n0 + wn + tn * 16 + li;
        float val = acc[tm][tn][v];
        if (OUT_F32)
          ((float*)Cout)[(size_t)m * N + n] = val;
        else
          ((bf16*)Cout)[(size_t)m * N + n] = (bf16)val;
      }
}

// ---------------------------------------------------------------------------
// Flash attention: block = (b, h, 64 q rows), 4 waves x 16 q rows each.
// ---------------------------------------------------------------------------
__global__ __launch_bounds__(128) void attn_kernel(
    const bf16* __restrict__ Q, const bf16* __restrict__ Kh,
    const bf16* __restrict__ V, const int* __restrict__ pos,
    bf16* __restrict__ O) {
  __shared__ __align__(32) bf16 sK[32 * 128];     // [k][d]
  __shared__ __align__(32) bf16 sVt[128 * 32];    // [d][k] (transposed)
  __shared__ __align__(32) bf16 sP[4][16 * 32];   // per-wave P tile

  const int tid  = threadIdx.x;
  const int lane = tid & 31;
  const int wid  = tid >> 5;
  const int li   = lane & 15;
  const int hi   = lane >> 4;
  const int bh   = blockIdx.y;
  const int b    = bh >> 4;
  const int h    = bh & 15;
  const int q0   = blockIdx.x * 64;
  const int qb   = q0 + wid * 16;
  const int posq_max = pos[b * SS + q0 + 63];
  const size_t baseBH = (size_t)b * SS * D_MODEL + (size_t)h * HEAD_DIM;

  // resident Q fragments: 16x128 as 4 A-fragments (16x32)
  v16bf qf[4];
#pragma unroll
  for (int dt = 0; dt < 4; ++dt) {
    const bf16* p = Q + baseBH + (size_t)(qb + li) * D_MODEL + dt * 32;
    qf[dt] = ld_frag(p + hi * 16, p + hi * 16 + 8);
  }
  int pq[8];
#pragma unroll
  for (int v = 0; v < 8; ++v) pq[v] = pos[b * SS + qb + v + 8 * hi];

  v8f acc[8] = {};
  float mrow[8], lrow[8];
#pragma unroll
  for (int v = 0; v < 8; ++v) { mrow[v] = -1e30f; lrow[v] = 0.f; }

  const int crow = tid >> 2;        // 0..31
  const int ccol = (tid & 3) * 32;  // 0..96

  for (int kc = 0; kc < SS; kc += 32) {
    // sorted positions: once the first key of a chunk is strictly in the
    // future of every query of this block, all later chunks are fully masked
    if (pos[b * SS + kc] > posq_max) break;
    __syncthreads();
    // stage K chunk [32][128] (contiguous -> async DMA when available)
    {
      const bf16* g = Kh + baseBH + (size_t)(kc + crow) * D_MODEL + ccol;
#if HAVE_ASYNC_LDS
#pragma unroll
      for (int j = 0; j < 4; ++j)
        async_cp16(&sK[crow * 128 + ccol + j * 8], g + j * 8);
#else
#pragma unroll
      for (int j = 0; j < 4; ++j)
        *(f32x4*)&sK[crow * 128 + ccol + j * 8] = *(const f32x4*)(g + j * 8);
#endif
    }
    // stage V chunk transposed -> sVt[d][k]
    {
      const bf16* g = V + baseBH + (size_t)(kc + crow) * D_MODEL + ccol;
      union { f32x4 f[4]; bf16 e[32]; } u;
#pragma unroll
      for (int j = 0; j < 4; ++j) u.f[j] = *(const f32x4*)(g + j * 8);
#pragma unroll
      for (int j = 0; j < 32; ++j) sVt[(ccol + j) * 32 + crow] = u.e[j];
    }
#if HAVE_ASYNC_LDS
    wait_async0();
#endif
    __syncthreads();

    // scores: two 16x16 tiles over keys kc..kc+15 and kc+16..kc+31
    v8f s0 = {}, s1 = {};
#pragma unroll
    for (int dt = 0; dt < 4; ++dt) {
      v16bf b0 = *(const v16bf*)&sK[li * 128 + dt * 32 + hi * 16];
      v16bf b1 = *(const v16bf*)&sK[(16 + li) * 128 + dt * 32 + hi * 16];
      s0 = wmma_bf16(qf[dt], b0, s0);
      s1 = wmma_bf16(qf[dt], b1, s1);
    }
    const float scale = 0.08838834764831845f;  // 1/sqrt(128)
    const int pk0 = pos[b * SS + kc + li];
    const int pk1 = pos[b * SS + kc + 16 + li];
    float p0[8], p1[8], mnew[8];
#pragma unroll
    for (int v = 0; v < 8; ++v) {
      float a0 = (pq[v] >= pk0) ? s0[v] * scale : -1e30f;
      float a1 = (pq[v] >= pk1) ? s1[v] * scale : -1e30f;
      p0[v] = a0; p1[v] = a1;
      float c = fmaxf(a0, a1);
#pragma unroll
      for (int msk = 1; msk <= 8; msk <<= 1)
        c = fmaxf(c, __shfl_xor(c, msk, 32));
      mnew[v] = fmaxf(mrow[v], c);
    }
#pragma unroll
    for (int v = 0; v < 8; ++v) {
      float alpha = __expf(mrow[v] - mnew[v]);
      float e0 = __expf(p0[v] - mnew[v]);   // masked -> underflows to 0
      float e1 = __expf(p1[v] - mnew[v]);
      float rs = e0 + e1;
#pragma unroll
      for (int msk = 1; msk <= 8; msk <<= 1)
        rs += __shfl_xor(rs, msk, 32);
      lrow[v] = lrow[v] * alpha + rs;
      mrow[v] = mnew[v];
#pragma unroll
      for (int t = 0; t < 8; ++t) acc[t][v] *= alpha;
      p0[v] = e0; p1[v] = e1;
    }
    // P (C-layout) -> LDS row-major 16x32, then reload in A-fragment layout
    bf16* sp = sP[wid];
#pragma unroll
    for (int v = 0; v < 8; ++v) {
      int m = v + 8 * hi;
      sp[m * 32 + li]      = (bf16)p0[v];
      sp[m * 32 + 16 + li] = (bf16)p1[v];
    }
    __syncthreads();
    v16bf pa = ld_frag(&sp[li * 32 + hi * 16], &sp[li * 32 + hi * 16 + 8]);
#pragma unroll
    for (int dt = 0; dt < 8; ++dt) {
      v16bf bv = *(const v16bf*)&sVt[(dt * 16 + li) * 32 + hi * 16];
      acc[dt] = wmma_bf16(pa, bv, acc[dt]);
    }
  }

  // normalize and write O (bf16, flattened [B*S, D_MODEL])
#pragma unroll
  for (int v = 0; v < 8; ++v) {
    float inv = (lrow[v] > 0.f) ? 1.f / lrow[v] : 0.f;
    int qr = qb + v + 8 * hi;
#pragma unroll
    for (int dt = 0; dt < 8; ++dt)
      O[baseBH + (size_t)qr * D_MODEL + dt * 16 + li] =
          (bf16)(acc[dt][v] * inv);
  }
}

// ---------------------------------------------------------------------------
// launch
// ---------------------------------------------------------------------------
extern "C" void kernel_launch(void* const* d_in, const int* in_sizes, int n_in,
                              void* d_out, int out_size, void* d_ws,
                              size_t ws_size, hipStream_t stream) {
  (void)in_sizes; (void)n_in; (void)out_size; (void)ws_size;
  const float* x  = (const float*)d_in[0];
  const float* Wq = (const float*)d_in[1];
  const float* Wk = (const float*)d_in[2];
  const float* Wv = (const float*)d_in[3];
  const float* Wo = (const float*)d_in[4];
  const int*  pos = (const int*)d_in[5];

  const size_t NX = (size_t)BB * SS * D_MODEL;      // 8,388,608
  const size_t NW = (size_t)D_MODEL * D_MODEL;      // 4,194,304
  char* ws = (char*)d_ws;
  bf16* xb  = (bf16*)ws; ws += NX * 2;
  bf16* wqb = (bf16*)ws; ws += NW * 2;
  bf16* wkb = (bf16*)ws; ws += NW * 2;
  bf16* wvb = (bf16*)ws; ws += NW * 2;
  bf16* wob = (bf16*)ws; ws += NW * 2;
  bf16* Qb  = (bf16*)ws; ws += NX * 2;
  bf16* Kb  = (bf16*)ws; ws += NX * 2;
  bf16* Vb  = (bf16*)ws; ws += NX * 2;
  bf16* Ob  = (bf16*)ws; ws += NX * 2;

  const int thr = 256;
  cvt_f32_bf16<<<(int)(NX / 4 + thr - 1) / thr, thr, 0, stream>>>(x,  xb,  (int)NX);
  cvt_f32_bf16<<<(int)(NW / 4 + thr - 1) / thr, thr, 0, stream>>>(Wq, wqb, (int)NW);
  cvt_f32_bf16<<<(int)(NW / 4 + thr - 1) / thr, thr, 0, stream>>>(Wk, wkb, (int)NW);
  cvt_f32_bf16<<<(int)(NW / 4 + thr - 1) / thr, thr, 0, stream>>>(Wv, wvb, (int)NW);
  cvt_f32_bf16<<<(int)(NW / 4 + thr - 1) / thr, thr, 0, stream>>>(Wo, wob, (int)NW);

  dim3 gg(D_MODEL / 128, (BB * SS) / 64);  // (16, 64)
  gemm_bf16k<false><<<gg, 128, 0, stream>>>(xb, wqb, Qb, BB * SS, D_MODEL, D_MODEL);
  gemm_bf16k<false><<<gg, 128, 0, stream>>>(xb, wkb, Kb, BB * SS, D_MODEL, D_MODEL);
  gemm_bf16k<false><<<gg, 128, 0, stream>>>(xb, wvb, Vb, BB * SS, D_MODEL, D_MODEL);

  const int rtot = BB * SS * N_HEADS * 64;
  rope_kernel<<<(rtot + 255) / 256, 256, 0, stream>>>(Qb, pos, rtot);
  rope_kernel<<<(rtot + 255) / 256, 256, 0, stream>>>(Kb, pos, rtot);

  attn_kernel<<<dim3(SS / 64, BB * N_HEADS), 128, 0, stream>>>(Qb, Kb, Vb, pos, Ob);

  gemm_bf16k<true><<<gg, 128, 0, stream>>>(Ob, wob, d_out, BB * SS, D_MODEL, D_MODEL);
}
